// LSTM_17978733101327
// MI455X (gfx1250) — matrix operand order repaired and checked by reading, not compiled, not run on hardware
//
#include <hip/hip_runtime.h>
#include <hip/hip_bf16.h>

// ---------------------------------------------------------------------------
// ConvLSTM2D x2 -> MaxPool3D -> Dense(10) -> Dense(1) for MI455X (gfx1250).
// All conv GEMMs run on v_wmma_f32_16x16x32_bf16 (bf16 A/B, f32 accumulate).
// One launch per recurrence step; h state double-buffered in zero-padded bf16.
// Workspace use: ~70 MB (fits MI455X 192MB L2).
// ---------------------------------------------------------------------------

typedef __attribute__((ext_vector_type(16))) __bf16 v16bf;
typedef __attribute__((ext_vector_type(8)))  float  v8f;
typedef __attribute__((ext_vector_type(4)))  unsigned u32x4;

union FragU { v16bf v; u32x4 q[2]; };

// A-fragment (16x32 bf16): lane<16 holds K=[0..7]&[16..23] of row lane,
// lane>=16 holds K=[8..15]&[24..31] of row lane-16. Caller passes p = base+lo.
__device__ __forceinline__ v16bf load_afrag(const __bf16* p) {
  FragU f;
  f.q[0] = *reinterpret_cast<const u32x4*>(p);
  f.q[1] = *reinterpret_cast<const u32x4*>(p + 16);
  return f.v;
}
// B-fragment from pre-packed buffer: 8 consecutive u32 per lane.
__device__ __forceinline__ v16bf load_bfrag(const unsigned* p) {
  FragU f;
  f.q[0] = *reinterpret_cast<const u32x4*>(p);
  f.q[1] = *reinterpret_cast<const u32x4*>(p + 4);
  return f.v;
}
__device__ __forceinline__ float hsig(float x) {
  return fminf(fmaxf(0.2f * x + 0.5f, 0.0f), 1.0f);
}

// ---------------------------------------------------------------------------
// Pack weight [K,128] f32 -> WMMA B-fragment layout, bf16, zero-pad K to 32*nkt.
// dst[((kt*8+nt)*32+lane)*8+j] = {bf16(W[K0,col]), bf16(W[K0+1,col])}
//   K0 = kt*32 + (lane<16?0:8) + 2j + (j<4?0:8), col = nt*16 + (lane&15)
// ---------------------------------------------------------------------------
__global__ __launch_bounds__(256) void pack_w_kernel(
    const float* __restrict__ src, unsigned* __restrict__ dst,
    int Ktot, int nkt)
{
  int i = blockIdx.x * blockDim.x + threadIdx.x;
  int tot = nkt * 2048;
  if (i >= tot) return;
  int j    = i & 7;
  int lane = (i >> 3) & 31;
  int nt   = (i >> 8) & 7;
  int kt   = i >> 11;
  int K0   = kt * 32 + ((lane < 16) ? 0 : 8) + 2 * j + ((j < 4) ? 0 : 8);
  int col  = nt * 16 + (lane & 15);
  float v0 = (K0     < Ktot) ? src[(size_t)K0 * 128 + col]       : 0.0f;
  float v1 = (K0 + 1 < Ktot) ? src[(size_t)(K0 + 1) * 128 + col] : 0.0f;
  union { __bf16 h[2]; unsigned u; } p;
  p.h[0] = (__bf16)v0;
  p.h[1] = (__bf16)v1;
  dst[i] = p.u;
}

// wEff[k] = sum_j Wd1[k,j] * Wd2[j]
__global__ __launch_bounds__(256) void weff_kernel(
    const float* __restrict__ Wd1, const float* __restrict__ Wd2,
    float* __restrict__ wEff)
{
  int k = blockIdx.x * blockDim.x + threadIdx.x;
  if (k >= 194304) return;
  float s = 0.0f;
  #pragma unroll
  for (int j = 0; j < 10; ++j) s += Wd1[(size_t)k * 10 + j] * Wd2[j];
  wEff[k] = s;
}

__global__ __launch_bounds__(256) void zero_kernel(unsigned* __restrict__ p, int n)
{
  int i = blockIdx.x * blockDim.x + threadIdx.x;
  if (i < n) p[i] = 0u;
}

// ---------------------------------------------------------------------------
// Layer-1 ConvLSTM step.
//   x: [16,12,48,48,3] f32,   gates z: [n=16*47*46, 128]
//   input conv VALID (K=18, one zero-padded K=32 WMMA step)
//   recurrent conv SAME via padded hin [16,48,48,32] bf16 (borders zero)
// Each wave: 16 rows x 128 cols of z -> gates -> c1, hout, h1seq.
// ---------------------------------------------------------------------------
__global__ __launch_bounds__(256) void step1_kernel(
    const float* __restrict__ x, const float* __restrict__ b1,
    const unsigned* __restrict__ W1p, const unsigned* __restrict__ U1p,
    const __bf16* __restrict__ hin, __bf16* __restrict__ hout,
    float* __restrict__ c1, __bf16* __restrict__ hseq, int tstep)
{
  const int lane = threadIdx.x & 31;
  const int wv   = threadIdx.x >> 5;
  const int tile = blockIdx.x * 8 + wv;
  if (tile >= 2162) return;                 // 16*47*46/16 row-tiles
  const int lo = (lane < 16) ? 0 : 8;

  // A-row owned by this lane
  int rowA = tile * 16 + (lane & 15);
  unsigned woA = (unsigned)rowA % 46u;
  unsigned tA  = (unsigned)rowA / 46u;
  unsigned hoA = tA % 47u;
  unsigned bA  = tA / 47u;

  v8f acc[8];
  #pragma unroll
  for (int i = 0; i < 8; ++i) {
    v8f z = {0.f, 0.f, 0.f, 0.f, 0.f, 0.f, 0.f, 0.f};
    acc[i] = z;
  }

  // ---- input conv: K=18 padded to one K=32 step; A built from f32 x ----
  {
    size_t xbase = ((((size_t)bA * 12 + tstep) * 48 + hoA) * 48 + woA) * 3;
    FragU af;
    #pragma unroll
    for (int j = 0; j < 8; ++j) {
      int k = lo + j;
      float v = 0.0f;
      if (k < 18) {
        int kh = k / 9, r9 = k % 9, kw = r9 / 3, ci = r9 % 3;
        v = x[xbase + (size_t)kh * 144 + kw * 3 + ci];
      }
      af.v[j] = (__bf16)v;
      int k2 = lo + 16 + j;
      float v2 = 0.0f;
      if (k2 < 18) {
        int kh = k2 / 9, r9 = k2 % 9, kw = r9 / 3, ci = r9 % 3;
        v2 = x[xbase + (size_t)kh * 144 + kw * 3 + ci];
      }
      af.v[8 + j] = (__bf16)v2;
    }
    #pragma unroll
    for (int nt = 0; nt < 8; ++nt) {
      v16bf bf = load_bfrag(W1p + (nt * 32 + lane) * 8);
      acc[nt] = __builtin_amdgcn_wmma_f32_16x16x32_bf16(
          false, af.v, false, bf, (short)0, acc[nt], false, false);
    }
  }

  // ---- recurrent conv: K=192 = 6 chunks; hin padded [16,48,48,32] ----
  #pragma unroll
  for (int kh = 0; kh < 2; ++kh) {
    size_t hbase = (((size_t)bA * 48 + hoA + kh) * 48 + woA) * 32;
    #pragma unroll
    for (int kc = 0; kc < 3; ++kc) {
      v16bf a = load_afrag(hin + hbase + kc * 32 + lo);
      const int kt = kh * 3 + kc;
      #pragma unroll
      for (int nt = 0; nt < 8; ++nt) {
        v16bf bf = load_bfrag(U1p + ((kt * 8 + nt) * 32 + lane) * 8);
        acc[nt] = __builtin_amdgcn_wmma_f32_16x16x32_bf16(
            false, a, false, bf, (short)0, acc[nt], false, false);
      }
    }
  }

  // ---- gates: lane holds rows {r or r+8}, cols {f, 16+f} per gate ----
  const int cl  = lane & 15;
  const int rsh = (lane < 16) ? 0 : 8;
  #pragma unroll
  for (int r = 0; r < 8; ++r) {
    int n = tile * 16 + r + rsh;
    unsigned wo = (unsigned)n % 46u;
    unsigned tt = (unsigned)n / 46u;
    unsigned ho = tt % 47u;
    unsigned bb = tt / 47u;
    #pragma unroll
    for (int hf = 0; hf < 2; ++hf) {
      int f = hf * 16 + cl;
      float zi = acc[0 + hf][r] + b1[f];
      float zf = acc[2 + hf][r] + b1[32 + f];
      float zc = acc[4 + hf][r] + b1[64 + f];
      float zo = acc[6 + hf][r] + b1[96 + f];
      size_t cidx = (size_t)n * 32 + f;
      float cold = c1[cidx];
      float cn = hsig(zf) * cold + hsig(zi) * fmaxf(zc, 0.0f);
      float hn = hsig(zo) * fmaxf(cn, 0.0f);
      c1[cidx] = cn;
      __bf16 hb = (__bf16)hn;
      hout[(((size_t)bb * 48 + ho) * 48 + (wo + 1)) * 32 + f] = hb;
      hseq[((((size_t)bb * 12 + tstep) * 47 + ho) * 46 + wo) * 32 + f] = hb;
    }
  }
}

// ---------------------------------------------------------------------------
// Layer-2 ConvLSTM step. Input conv VALID over h1seq[:,t] [47,46,32] bf16
// (K=192), recurrent SAME via padded hin [16,47,46,32]. Output 46x44x(4*32).
// ---------------------------------------------------------------------------
__global__ __launch_bounds__(256) void step2_kernel(
    const __bf16* __restrict__ h1seq, const float* __restrict__ b2,
    const unsigned* __restrict__ W2p, const unsigned* __restrict__ U2p,
    const __bf16* __restrict__ hin, __bf16* __restrict__ hout,
    float* __restrict__ c2, __bf16* __restrict__ h2seq, int tstep)
{
  const int lane = threadIdx.x & 31;
  const int wv   = threadIdx.x >> 5;
  const int tile = blockIdx.x * 8 + wv;
  if (tile >= 2024) return;                 // 16*46*44/16
  const int lo = (lane < 16) ? 0 : 8;

  int rowA = tile * 16 + (lane & 15);
  unsigned woA = (unsigned)rowA % 44u;
  unsigned tA  = (unsigned)rowA / 44u;
  unsigned hoA = tA % 46u;
  unsigned bA  = tA / 46u;

  v8f acc[8];
  #pragma unroll
  for (int i = 0; i < 8; ++i) {
    v8f z = {0.f, 0.f, 0.f, 0.f, 0.f, 0.f, 0.f, 0.f};
    acc[i] = z;
  }

  // ---- input conv (valid) from h1seq [16,12,47,46,32] ----
  #pragma unroll
  for (int kh = 0; kh < 2; ++kh) {
    size_t ib = ((((size_t)bA * 12 + tstep) * 47 + hoA + kh) * 46 + woA) * 32;
    #pragma unroll
    for (int kc = 0; kc < 3; ++kc) {
      v16bf a = load_afrag(h1seq + ib + kc * 32 + lo);
      const int kt = kh * 3 + kc;
      #pragma unroll
      for (int nt = 0; nt < 8; ++nt) {
        v16bf bf = load_bfrag(W2p + ((kt * 8 + nt) * 32 + lane) * 8);
        acc[nt] = __builtin_amdgcn_wmma_f32_16x16x32_bf16(
            false, a, false, bf, (short)0, acc[nt], false, false);
      }
    }
  }
  // ---- recurrent conv (same) from padded hin [16,47,46,32] ----
  #pragma unroll
  for (int kh = 0; kh < 2; ++kh) {
    size_t hb = (((size_t)bA * 47 + hoA + kh) * 46 + woA) * 32;
    #pragma unroll
    for (int kc = 0; kc < 3; ++kc) {
      v16bf a = load_afrag(hin + hb + kc * 32 + lo);
      const int kt = kh * 3 + kc;
      #pragma unroll
      for (int nt = 0; nt < 8; ++nt) {
        v16bf bf = load_bfrag(U2p + ((kt * 8 + nt) * 32 + lane) * 8);
        acc[nt] = __builtin_amdgcn_wmma_f32_16x16x32_bf16(
            false, a, false, bf, (short)0, acc[nt], false, false);
      }
    }
  }

  // ---- gates ----
  const int cl  = lane & 15;
  const int rsh = (lane < 16) ? 0 : 8;
  #pragma unroll
  for (int r = 0; r < 8; ++r) {
    int n = tile * 16 + r + rsh;
    unsigned wo = (unsigned)n % 44u;
    unsigned tt = (unsigned)n / 44u;
    unsigned ho = tt % 46u;
    unsigned bb = tt / 46u;
    #pragma unroll
    for (int hf = 0; hf < 2; ++hf) {
      int f = hf * 16 + cl;
      float zi = acc[0 + hf][r] + b2[f];
      float zf = acc[2 + hf][r] + b2[32 + f];
      float zc = acc[4 + hf][r] + b2[64 + f];
      float zo = acc[6 + hf][r] + b2[96 + f];
      size_t cidx = (size_t)n * 32 + f;
      float cold = c2[cidx];
      float cn = hsig(zf) * cold + hsig(zi) * fmaxf(zc, 0.0f);
      float hn = hsig(zo) * fmaxf(cn, 0.0f);
      c2[cidx] = cn;
      __bf16 hb = (__bf16)hn;
      hout[(((size_t)bb * 47 + ho) * 46 + (wo + 1)) * 32 + f] = hb;
      h2seq[((((size_t)bb * 12 + tstep) * 46 + ho) * 44 + wo) * 32 + f] = hb;
    }
  }
}

// ---------------------------------------------------------------------------
// MaxPool3D(1,2,2) over h2seq [16,12,46,44,32] fused with the collapsed
// dense head: out[b] = sum_k pool[b,k]*wEff[k] + (bd1.Wd2 + bd2).
// One block per batch element.
// ---------------------------------------------------------------------------
__global__ __launch_bounds__(256) void final_kernel(
    const __bf16* __restrict__ h2seq, const float* __restrict__ wEff,
    const float* __restrict__ bd1, const float* __restrict__ Wd2,
    const float* __restrict__ bd2, float* __restrict__ out)
{
  const int b = blockIdx.x;
  const int tid = threadIdx.x;
  float s = 0.0f;
  for (int k = tid; k < 194304; k += 256) {
    int f   = k & 31;
    int rem = k >> 5;
    int pw  = rem % 22; rem /= 22;
    int ph  = rem % 23;
    int tt  = rem / 23;
    size_t base = ((((size_t)b * 12 + tt) * 46 + 2 * ph) * 44 + 2 * pw) * 32 + f;
    float m = (float)h2seq[base];
    m = fmaxf(m, (float)h2seq[base + 32]);
    m = fmaxf(m, (float)h2seq[base + 44 * 32]);
    m = fmaxf(m, (float)h2seq[base + 44 * 32 + 32]);
    s += m * wEff[k];
  }
  __shared__ float red[256];
  red[tid] = s;
  __syncthreads();
  for (int off = 128; off > 0; off >>= 1) {
    if (tid < off) red[tid] += red[tid + off];
    __syncthreads();
  }
  if (tid == 0) {
    float be = bd2[0];
    #pragma unroll
    for (int j = 0; j < 10; ++j) be += bd1[j] * Wd2[j];
    out[b] = red[0] + be;
  }
}

// ---------------------------------------------------------------------------
extern "C" void kernel_launch(void* const* d_in, const int* in_sizes, int n_in,
                              void* d_out, int out_size, void* d_ws, size_t ws_size,
                              hipStream_t stream)
{
  (void)in_sizes; (void)n_in; (void)out_size; (void)ws_size;
  const float* x   = (const float*)d_in[0];
  const float* W1  = (const float*)d_in[1];
  const float* U1  = (const float*)d_in[2];
  const float* b1  = (const float*)d_in[3];
  const float* W2  = (const float*)d_in[4];
  const float* U2  = (const float*)d_in[5];
  const float* b2  = (const float*)d_in[6];
  const float* Wd1 = (const float*)d_in[7];
  const float* bd1 = (const float*)d_in[8];
  const float* Wd2 = (const float*)d_in[9];
  const float* bd2 = (const float*)d_in[10];
  float* out = (float*)d_out;

  char* ws = (char*)d_ws;
  size_t off = 0;
  auto wsalloc = [&](size_t bytes) -> void* {
    void* p = ws + off;
    off += (bytes + 255) & ~(size_t)255;
    return p;
  };
  unsigned* W1p  = (unsigned*)wsalloc((size_t)2048 * 4);       // K pad 32, 1 ktile
  unsigned* U1p  = (unsigned*)wsalloc((size_t)12288 * 4);      // K=192, 6 ktiles
  unsigned* W2p  = (unsigned*)wsalloc((size_t)12288 * 4);
  unsigned* U2p  = (unsigned*)wsalloc((size_t)12288 * 4);
  float*    wEff = (float*)wsalloc((size_t)194304 * 4);
  __bf16* hp1a   = (__bf16*)wsalloc((size_t)16 * 48 * 48 * 32 * 2);
  __bf16* hp1b   = (__bf16*)wsalloc((size_t)16 * 48 * 48 * 32 * 2);
  float*  c1     = (float*)wsalloc((size_t)16 * 47 * 46 * 32 * 4);
  __bf16* h1seq  = (__bf16*)wsalloc((size_t)16 * 12 * 47 * 46 * 32 * 2);
  __bf16* hp2a   = (__bf16*)wsalloc((size_t)16 * 47 * 46 * 32 * 2);
  __bf16* hp2b   = (__bf16*)wsalloc((size_t)16 * 47 * 46 * 32 * 2);
  float*  c2     = (float*)wsalloc((size_t)16 * 46 * 44 * 32 * 4);
  __bf16* h2seq  = (__bf16*)wsalloc((size_t)16 * 12 * 46 * 44 * 32 * 2);

  // Pack weights into WMMA B-fragment bf16 layout.
  pack_w_kernel<<<8,  256, 0, stream>>>(W1, W1p, 18, 1);
  pack_w_kernel<<<48, 256, 0, stream>>>(U1, U1p, 192, 6);
  pack_w_kernel<<<48, 256, 0, stream>>>(W2, W2p, 192, 6);
  pack_w_kernel<<<48, 256, 0, stream>>>(U2, U2p, 192, 6);
  // Collapse dense head: wEff = Wd1 @ Wd2.
  weff_kernel<<<759, 256, 0, stream>>>(Wd1, Wd2, wEff);
  // Zero state (padded h buffers incl. borders, cell states).
  zero_kernel<<<2304, 256, 0, stream>>>((unsigned*)hp1a, 589824);
  zero_kernel<<<2304, 256, 0, stream>>>((unsigned*)hp1b, 589824);
  zero_kernel<<<4322, 256, 0, stream>>>((unsigned*)c1,   1106432);
  zero_kernel<<<2161, 256, 0, stream>>>((unsigned*)hp2a, 553216);
  zero_kernel<<<2161, 256, 0, stream>>>((unsigned*)hp2b, 553216);
  zero_kernel<<<4048, 256, 0, stream>>>((unsigned*)c2,   1036288);

  // Layer-1 recurrence (double-buffered padded h to avoid intra-step races).
  for (int t = 0; t < 12; ++t) {
    const __bf16* hin = (t & 1) ? hp1b : hp1a;
    __bf16* hout      = (t & 1) ? hp1a : hp1b;
    step1_kernel<<<271, 256, 0, stream>>>(x, b1, W1p, U1p, hin, hout, c1, h1seq, t);
  }
  // Layer-2 recurrence.
  for (int t = 0; t < 12; ++t) {
    const __bf16* hin = (t & 1) ? hp2b : hp2a;
    __bf16* hout      = (t & 1) ? hp2a : hp2b;
    step2_kernel<<<253, 256, 0, stream>>>(h1seq, b2, W2p, U2p, hin, hout, c2, h2seq, t);
  }
  // Pool + dense head.
  final_kernel<<<16, 256, 0, stream>>>(h2seq, wEff, bd1, Wd2, bd2, out);
}
